// AutoformerEncoderBlock_50921132261792
// MI455X (gfx1250) — compile-verified, matrix operand order checked by reading, not run on hardware
//
#include <hip/hip_runtime.h>
#include <hip/hip_bf16.h>
#include <math.h>

// Problem constants (Autoformer encoder block)
#define B_   8
#define S_   1024
#define D_   512
#define H_   8
#define DP_  64
#define KWIN 25
#define EPSV 1e-6f

typedef unsigned short u16;
typedef unsigned int   u32;
typedef __attribute__((ext_vector_type(16))) __bf16 v16bf;
typedef __attribute__((ext_vector_type(8)))  float  v8f;
typedef __attribute__((ext_vector_type(4)))  unsigned uint32x4;
typedef __attribute__((ext_vector_type(8)))  int      int32x8;
typedef __attribute__((ext_vector_type(4)))  int      int32x4;

// ---- feature detection: TDM builtins (arity differs across toolchains) ----
#if defined(__has_builtin)
# if __has_builtin(__builtin_amdgcn_tensor_load_to_lds) && __has_builtin(__builtin_amdgcn_s_wait_tensorcnt)
#  define USE_TDM 1
# endif
#endif
#ifndef USE_TDM
# define USE_TDM 0
#endif
#if defined(__has_include)
# if __has_include(<hip/amd_detail/amd_gfx1250_TDM.h>)
#  define TDM_ARITY6 1
# endif
#endif
#ifndef TDM_ARITY6
# define TDM_ARITY6 0
#endif

__device__ __forceinline__ u16 f2bf(float f) {
  unsigned u = __float_as_uint(f);
  u += 0x7FFFu + ((u >> 16) & 1u);   // round-to-nearest-even
  return (u16)(u >> 16);
}

union FragU { v16bf v; u16 u[16]; };

#if USE_TDM
// ---------------------------------------------------------------------------
// Issue one TDM load of a [128 rows x 32 bf16] tile (row stride Kd elements)
// into LDS at lds_off, padding +4 DWORDs after each 16 DWORDs (64B row) so the
// LDS image has row stride 80B == u16[40] (bank-conflict-free frag reads).
// D# bitfields per CDNA5 ISA 8.3-8.6.
// ---------------------------------------------------------------------------
__device__ __forceinline__ void tdm_load_tile(const u16* gsrc, unsigned lds_off, int Kd) {
  unsigned long long ga = (unsigned long long)(uintptr_t)gsrc;
  uint32x4 g0;
  g0[0] = 1u;                                           // count=1, gather off
  g0[1] = lds_off;                                      // lds_addr (bytes)
  g0[2] = (unsigned)ga;                                 // global_addr[31:0]
  g0[3] = ((unsigned)(ga >> 32) & 0x01FFFFFFu)          // global_addr[56:32]
          | (2u << 30);                                 // type=2 ("image")
  int32x8 g1;
  g1[0] = (int)(0x00010000u      // data_size=1 (2 bytes)
              | 0x00100000u      // pad_enable
              | (3u << 22)       // pad_interval: 16 DWORDs (one 64B row)
              | (3u << 25));     // pad_amount:   4 DWORDs (16B) -> 80B stride
  g1[1] = (int)(((unsigned)Kd & 0xFFFFu) << 16);        // tensor_dim0[15:0]
  g1[2] = (int)(((unsigned)Kd >> 16) | (128u << 16));   // dim0 hi | tensor_dim1=128
  g1[3] = (int)(32u << 16);                             // tile_dim0 = 32 elems
  g1[4] = 128;                                          // tile_dim1 = 128 rows
  g1[5] = Kd;                                           // tensor_dim0_stride lo
  g1[6] = 0;
  g1[7] = 0;
  int32x4 g2; g2[0] = 1; g2[1] = 1; g2[2] = 0; g2[3] = 0;   // tensor_dim2/3 = 1
  int32x4 g3; g3[0] = 0; g3[1] = (int)(1u << 16); g3[2] = 0; g3[3] = 0; // dim4=1
#if TDM_ARITY6
  int32x8 gz; for (int i = 0; i < 8; ++i) gz[i] = 0;
  __builtin_amdgcn_tensor_load_to_lds(g0, g1, g2, g3, gz, 0);
#else
  __builtin_amdgcn_tensor_load_to_lds(g0, g1, g2, g3, 0);
#endif
}
#endif

// ---------------------------------------------------------------------------
// Tiled bf16 WMMA GEMM: C[M,N] = A[M,K] @ B[K,N] + bias, fp32 accumulate.
// A: bf16 [M,K] row-major.  Bt: bf16 [N,K] row-major (pre-transposed).
// Block: 256 threads (8 wave32), tile 128x128, K-step 32; wave = 64x32 out.
// TDM path: double-buffered LDS; pair(i+1) in flight while pair(i) computes
// (tensor ops of one wave complete in order -> s_wait_tensorcnt 2 gates pair i).
// EPI: 0 = +bias (f32 out), 1 = +bias+residual (f32 out), 2 = +bias,ReLU (bf16 out)
// ---------------------------------------------------------------------------
template<int EPI>
__global__ __launch_bounds__(256) void gemm_wmma(
    const u16* __restrict__ A, const u16* __restrict__ Bt,
    const float* __restrict__ bias, const float* __restrict__ res,
    void* __restrict__ Cv, int M, int N, int Kd)
{
#if USE_TDM
  __shared__ u16 As[2][128][40];   // [buf][m][k], 80B row stride (TDM pad)
  __shared__ u16 Bs[2][128][40];   // [buf][n][k]
#else
  __shared__ u16 As[1][128][40];
  __shared__ u16 Bs[1][128][40];
#endif

  const int tid  = threadIdx.x;
  const int m0   = blockIdx.y * 128;
  const int n0   = blockIdx.x * 128;
  const int wave = tid >> 5, lane = tid & 31;
  const int wm   = wave >> 2, wn = wave & 3;
  const int lrow = lane & 15, lhalf = (lane >> 4) << 4;
  const int nk   = Kd >> 5;

  v8f acc[4][2];
  #pragma unroll
  for (int i = 0; i < 4; ++i)
    #pragma unroll
    for (int j = 0; j < 2; ++j)
      #pragma unroll
      for (int c = 0; c < 8; ++c) acc[i][j][c] = 0.0f;

#if USE_TDM
  const unsigned ldsA0 = (unsigned)(uintptr_t)(void*)&As[0][0][0];
  const unsigned ldsA1 = (unsigned)(uintptr_t)(void*)&As[1][0][0];
  const unsigned ldsB0 = (unsigned)(uintptr_t)(void*)&Bs[0][0][0];
  const unsigned ldsB1 = (unsigned)(uintptr_t)(void*)&Bs[1][0][0];
  const u16* Abase = A  + (size_t)m0 * Kd;
  const u16* Bbase = Bt + (size_t)n0 * Kd;

  if (tid < 32) {                       // wave 0 drives the Tensor Data Mover
    tdm_load_tile(Abase, ldsA0, Kd);
    tdm_load_tile(Bbase, ldsB0, Kd);
    if (nk > 1) {
      tdm_load_tile(Abase + 32, ldsA1, Kd);
      tdm_load_tile(Bbase + 32, ldsB1, Kd);
    }
  }
#endif

  for (int i = 0; i < nk; ++i) {
    const int bsel = i & 1;
#if USE_TDM
    if (tid < 32) {
      if (i + 1 < nk) __builtin_amdgcn_s_wait_tensorcnt(2);  // pair i landed
      else            __builtin_amdgcn_s_wait_tensorcnt(0);  // last pair
    }
#else
    {
      const int srow = tid >> 1, shalf = (tid & 1) << 4;     // 16 u16 per thread
      const u32* ga = (const u32*)(A  + (size_t)(m0 + srow) * Kd + i*32 + shalf);
      const u32* gb = (const u32*)(Bt + (size_t)(n0 + srow) * Kd + i*32 + shalf);
      u32* da = (u32*)&As[0][srow][shalf];
      u32* db = (u32*)&Bs[0][srow][shalf];
      #pragma unroll
      for (int q = 0; q < 8; ++q) da[q] = ga[q];
      #pragma unroll
      for (int q = 0; q < 8; ++q) db[q] = gb[q];
    }
#endif
    __syncthreads();

    FragU af[4], bfr[2];
    #pragma unroll
    for (int ii = 0; ii < 4; ++ii) {
      const u16* src = &As[bsel][wm*64 + ii*16 + lrow][lhalf];
      #pragma unroll
      for (int c = 0; c < 16; ++c) af[ii].u[c] = src[c];
    }
    #pragma unroll
    for (int jj = 0; jj < 2; ++jj) {
      const u16* src = &Bs[bsel][wn*32 + jj*16 + lrow][lhalf];
      #pragma unroll
      for (int c = 0; c < 16; ++c) bfr[jj].u[c] = src[c];
    }
    #pragma unroll
    for (int ii = 0; ii < 4; ++ii)
      #pragma unroll
      for (int jj = 0; jj < 2; ++jj)
        acc[ii][jj] = __builtin_amdgcn_wmma_f32_16x16x32_bf16(
            false, af[ii].v, false, bfr[jj].v, (short)0, acc[ii][jj], false, false);
    __syncthreads();                    // all waves done reading buf bsel

#if USE_TDM
    if (tid < 32 && i + 2 < nk) {       // refill the buffer just consumed
      tdm_load_tile(Abase + (size_t)(i + 2) * 32, bsel ? ldsA1 : ldsA0, Kd);
      tdm_load_tile(Bbase + (size_t)(i + 2) * 32, bsel ? ldsB1 : ldsB0, Kd);
    }
#endif
  }

  // epilogue: VGPR r holds row r (lanes 0-15) / row 8+r (lanes 16-31)
  const int rh = (lane >> 4) * 8, cl = lane & 15;
  #pragma unroll
  for (int i = 0; i < 4; ++i)
    #pragma unroll
    for (int j = 0; j < 2; ++j)
      #pragma unroll
      for (int r = 0; r < 8; ++r) {
        int gm = m0 + wm*64 + i*16 + rh + r;
        int gn = n0 + wn*32 + j*16 + cl;
        float v = acc[i][j][r] + bias[gn];
        if (EPI == 1) v += res[(size_t)gm * N + gn];
        if (EPI == 2) {
          v = fmaxf(v, 0.0f);
          ((u16*)Cv)[(size_t)gm * N + gn] = f2bf(v);
        } else {
          ((float*)Cv)[(size_t)gm * N + gn] = v;
        }
      }
}

// ---------------------------------------------------------------------------
// Weight convert + transpose: wt[n][k] = bf16(w[k][n]).  32x32 LDS tiles.
// ---------------------------------------------------------------------------
__global__ __launch_bounds__(256) void wconv_kernel(
    const float* __restrict__ w, u16* __restrict__ wt, int K, int N)
{
  __shared__ float tile[32][33];
  const int nb = blockIdx.x * 32, kb = blockIdx.y * 32;
  const int tx = threadIdx.x & 31, ty = threadIdx.x >> 5;
  #pragma unroll
  for (int r = ty; r < 32; r += 8)
    tile[r][tx] = w[(size_t)(kb + r) * N + nb + tx];
  __syncthreads();
  #pragma unroll
  for (int r = ty; r < 32; r += 8)
    wt[(size_t)(nb + r) * K + kb + tx] = f2bf(tile[tx][r]);
}

// ---------------------------------------------------------------------------
// Moving average ('SAME', divisor excludes padding); seasonal = x - trend
// (fp32 for residual paths + bf16 copy for the QKV GEMM A operand).
// ---------------------------------------------------------------------------
__global__ __launch_bounds__(256) void mavg_kernel(
    const float* __restrict__ x, float* __restrict__ trend,
    float* __restrict__ seasonal, u16* __restrict__ seasonal_bf)
{
  const int d  = blockIdx.x * 256 + threadIdx.x;   // grid.x = 2
  const int b  = blockIdx.y;
  const int s0 = blockIdx.z * 128;                 // grid.z = 8
  const float* xb = x + (size_t)b * S_ * D_ + d;

  float sum = 0.0f; int cnt = 0;
  int lo = s0 - KWIN/2; if (lo < 0) lo = 0;
  int hi = s0 + KWIN/2; if (hi > S_-1) hi = S_-1;
  for (int j = lo; j <= hi; ++j) { sum += xb[(size_t)j * D_]; ++cnt; }

  for (int s = s0; s < s0 + 128; ++s) {
    float t = sum / (float)cnt;
    size_t idx = ((size_t)b * S_ + s) * D_ + d;
    float se = x[idx] - t;
    trend[idx]       = t;
    seasonal[idx]    = se;
    seasonal_bf[idx] = f2bf(se);
    int add = s + 1 + KWIN/2;
    int rem = s - KWIN/2;
    if (add <= S_-1) { sum += xb[(size_t)add * D_]; ++cnt; }
    if (rem >= 0)    { sum -= xb[(size_t)rem * D_]; --cnt; }
  }
}

// Vrest[b, h*64+d] = sum_{t=64..S-1} V[b,t,h*64+d]
__global__ __launch_bounds__(256) void vrest_kernel(
    const float* __restrict__ V, float* __restrict__ vrest)
{
  const int d = blockIdx.x * 256 + threadIdx.x;
  const int b = blockIdx.y;
  const float* vb = V + (size_t)b * S_ * D_ + d;
  float s = 0.0f;
  for (int t = DP_; t < S_; ++t) s += vb[(size_t)t * D_];
  vrest[b * D_ + d] = s;
}

// ---------------------------------------------------------------------------
// Autocorrelation "attention": corr[t]=Σ_{n≥t} q[n]k[n-t] (0 for t>=64),
// softmax over 64 scores + 960 zeros, weighted sum of V rows 0..63 + Vrest.
// One block (64 threads) per (b,h,s). Output bf16 (A operand of wo GEMM).
// ---------------------------------------------------------------------------
__global__ __launch_bounds__(64) void autocorr_kernel(
    const float* __restrict__ Q, const float* __restrict__ Km,
    const float* __restrict__ V, const float* __restrict__ vrest,
    u16* __restrict__ outc)
{
  __shared__ float qs[64], ks[64], ws_[64], red[64];
  const int id = blockIdx.x;
  const int s = id & (S_-1);
  const int h = (id >> 10) & (H_-1);
  const int b = id >> 13;
  const int t = threadIdx.x;

  const size_t rowq = ((size_t)b * S_ + s) * D_ + h * DP_;
  qs[t] = Q[rowq + t];
  ks[t] = Km[rowq + t];
  __syncthreads();

  float c = 0.0f;
  for (int n = t; n < DP_; ++n) c += qs[n] * ks[n - t];
  float z = 0.125f * c;                       // scale = 1/sqrt(64)

  red[t] = z; __syncthreads();
  for (int st = 32; st > 0; st >>= 1) {
    if (t < st) red[t] = fmaxf(red[t], red[t + st]);
    __syncthreads();
  }
  float m = fmaxf(red[0], 0.0f);              // zero scores join the softmax
  __syncthreads();

  float w = __expf(z - m);
  ws_[t] = w; red[t] = w; __syncthreads();
  for (int st = 32; st > 0; st >>= 1) {
    if (t < st) red[t] += red[t + st];
    __syncthreads();
  }
  float w0 = __expf(-m);
  float denom = red[0] + 960.0f * w0;

  const float* vb = V + (size_t)b * S_ * D_ + h * DP_ + t;
  float acc = w0 * vrest[b * D_ + h * DP_ + t];
  #pragma unroll 4
  for (int j = 0; j < DP_; ++j) acc += ws_[j] * vb[(size_t)j * D_];
  outc[rowq + t] = f2bf(acc / denom);
}

// ---------------------------------------------------------------------------
// LayerNorm over D=512: out = LN(in (+pre)) * g + be (+post); optional bf16 copy.
// ---------------------------------------------------------------------------
__global__ __launch_bounds__(256) void ln_kernel(
    const float* __restrict__ in, const float* __restrict__ pre,
    const float* __restrict__ g, const float* __restrict__ be,
    const float* __restrict__ post, float* __restrict__ out,
    u16* __restrict__ out_bf)
{
  __shared__ float r1[256], r2[256];
  const int row = blockIdx.x, t = threadIdx.x;
  const size_t base = (size_t)row * D_;

  float v0 = in[base + t];
  float v1 = in[base + t + 256];
  if (pre) { v0 += pre[base + t]; v1 += pre[base + t + 256]; }
  r1[t] = v0 + v1;
  r2[t] = v0*v0 + v1*v1;
  __syncthreads();
  for (int st = 128; st > 0; st >>= 1) {
    if (t < st) { r1[t] += r1[t+st]; r2[t] += r2[t+st]; }
    __syncthreads();
  }
  float mean = r1[0] * (1.0f / D_);
  float var  = r2[0] * (1.0f / D_) - mean * mean;
  float rs   = rsqrtf(var + EPSV);
  float o0 = (v0 - mean) * rs * g[t]       + be[t];
  float o1 = (v1 - mean) * rs * g[t + 256] + be[t + 256];
  if (post) { o0 += post[base + t]; o1 += post[base + t + 256]; }
  out[base + t]       = o0;
  out[base + t + 256] = o1;
  if (out_bf) { out_bf[base + t] = f2bf(o0); out_bf[base + t + 256] = f2bf(o1); }
}

// ---------------------------------------------------------------------------
extern "C" void kernel_launch(void* const* d_in, const int* in_sizes, int n_in,
                              void* d_out, int out_size, void* d_ws, size_t ws_size,
                              hipStream_t stream)
{
  const float* x  = (const float*)d_in[0];
  const float* wq = (const float*)d_in[1];  const float* bq = (const float*)d_in[2];
  const float* wk = (const float*)d_in[3];  const float* bk = (const float*)d_in[4];
  const float* wv = (const float*)d_in[5];  const float* bv = (const float*)d_in[6];
  const float* wo = (const float*)d_in[7];  const float* bo = (const float*)d_in[8];
  const float* w1 = (const float*)d_in[9];  const float* b1 = (const float*)d_in[10];
  const float* w2 = (const float*)d_in[11]; const float* b2 = (const float*)d_in[12];
  const float* g1 = (const float*)d_in[13]; const float* be1 = (const float*)d_in[14];
  const float* g2 = (const float*)d_in[15]; const float* be2 = (const float*)d_in[16];
  const float* g3 = (const float*)d_in[17]; const float* be3 = (const float*)d_in[18];
  (void)in_sizes; (void)n_in; (void)out_size; (void)ws_size;

  float* ws = (float*)d_ws;
  const size_t NE = (size_t)B_ * S_ * D_;          // 4,194,304
  const size_t WW = (size_t)D_ * D_;               // 262,144

  float* seasonal = ws + 0 * NE;
  float* trend    = ws + 1 * NE;
  float* y1       = ws + 2 * NE;
  float* out1     = ws + 3 * NE;
  float* y2       = ws + 4 * NE;
  float* sout     = ws + 5 * NE;
  float* Qb       = ws + 6 * NE;
  float* Kb       = ws + 7 * NE;
  float* Vb       = ws + 8 * NE;
  float* vrest    = ws + 9 * NE;                   // B*D floats

  u16* ub          = (u16*)(ws + 10 * NE);
  u16* seasonal_bf = ub;                           // NE
  u16* attnc_bf    = ub + 1 * NE;                  // NE
  u16* out1_bf     = ub + 2 * NE;                  // NE
  u16* hbuf_bf     = ub + 3 * NE;                  // 4*NE
  u16* wq_t        = ub + 7 * NE;
  u16* wk_t        = wq_t + WW;
  u16* wv_t        = wk_t + WW;
  u16* wo_t        = wv_t + WW;
  u16* w1_t        = wo_t + WW;                    // [2048][512]
  u16* w2_t        = w1_t + 4 * WW;                // [512][2048]

  const int M = B_ * S_;                           // 8192

  // 0) one-time weight convert + transpose to bf16 [N][K]
  wconv_kernel<<<dim3(D_/32,     D_/32),     256, 0, stream>>>(wq, wq_t, D_,   D_);
  wconv_kernel<<<dim3(D_/32,     D_/32),     256, 0, stream>>>(wk, wk_t, D_,   D_);
  wconv_kernel<<<dim3(D_/32,     D_/32),     256, 0, stream>>>(wv, wv_t, D_,   D_);
  wconv_kernel<<<dim3(D_/32,     D_/32),     256, 0, stream>>>(wo, wo_t, D_,   D_);
  wconv_kernel<<<dim3((4*D_)/32, D_/32),     256, 0, stream>>>(w1, w1_t, D_,   4*D_);
  wconv_kernel<<<dim3(D_/32,     (4*D_)/32), 256, 0, stream>>>(w2, w2_t, 4*D_, D_);

  // 1) decomposition (single moving average; trend2 cancels algebraically)
  mavg_kernel<<<dim3(D_/256, B_, S_/128), 256, 0, stream>>>(x, trend, seasonal, seasonal_bf);

  // 2) Q/K/V projections (bf16 WMMA, fp32 accum, double-buffered TDM tiles)
  dim3 gqkv(D_/128, M/128);                        // (4, 64)
  gemm_wmma<0><<<gqkv, 256, 0, stream>>>(seasonal_bf, wq_t, bq, nullptr, Qb, M, D_, D_);
  gemm_wmma<0><<<gqkv, 256, 0, stream>>>(seasonal_bf, wk_t, bk, nullptr, Kb, M, D_, D_);
  gemm_wmma<0><<<gqkv, 256, 0, stream>>>(seasonal_bf, wv_t, bv, nullptr, Vb, M, D_, D_);

  // 3) autocorrelation core (only lags 0..63 are nonzero)
  vrest_kernel<<<dim3(D_/256, B_), 256, 0, stream>>>(Vb, vrest);
  autocorr_kernel<<<B_ * H_ * S_, 64, 0, stream>>>(Qb, Kb, Vb, vrest, attnc_bf);

  // 4) output proj + residual, LN1 (also emits bf16 out1 for the FFN A)
  gemm_wmma<1><<<gqkv, 256, 0, stream>>>(attnc_bf, wo_t, bo, seasonal, y1, M, D_, D_);
  ln_kernel<<<M, 256, 0, stream>>>(y1, nullptr, g1, be1, nullptr, out1, out1_bf);

  // 5) FFN: relu(out1@w1+b1) -> bf16 hidden; @w2+b2 + out1; LN2 + seasonal
  gemm_wmma<2><<<dim3((4*D_)/128, M/128), 256, 0, stream>>>(out1_bf, w1_t, b1, nullptr, hbuf_bf, M, 4*D_, D_);
  gemm_wmma<1><<<gqkv, 256, 0, stream>>>(hbuf_bf, w2_t, b2, out1, y2, M, D_, 4*D_);
  ln_kernel<<<M, 256, 0, stream>>>(y2, nullptr, g2, be2, seasonal, sout, nullptr);

  // 6) x_out = sout + trend (trend2 cancels), final LN3
  ln_kernel<<<M, 256, 0, stream>>>(sout, trend, g3, be3, nullptr, (float*)d_out, nullptr);
}
